// GlobalAttentionPool_75453985456260
// MI455X (gfx1250) — matrix operand order repaired and checked by reading, not compiled, not run on hardware
//
#include <hip/hip_runtime.h>

typedef __attribute__((ext_vector_type(16))) _Float16 v16h;
typedef __attribute__((ext_vector_type(8)))  float    v8f;

#define DIM 128
#define EPS 1e-16f
#define NEG_INF (-__builtin_huge_valf())

// Fused GlobalAttentionPool, single pass over x.
// One block per segment (batch is sorted -> segments are contiguous), 128 threads
// = 4 waves. Each wave streams 32-node chunks with an online-softmax state
// (running max m_w, running denom s_l, 8x v8f WMMA accumulators holding the
// running weighted feature sums). Chunk pipeline:
//   1. lane computes score = x[node] . W + b   (row in float4, W from LDS)
//   2. wave max -> maybe rescale (m, s, acc) by exp(m_old - m_new)  [uniform]
//   3. p = exp(score - m)  -> A matrix (16x32 f16, rows replicated)
//   4. 8 feature groups: B tile (32x16 f16) -> v_wmma_f32_16x16x32_f16 chain
// Waves merge partials in LDS with per-wave scale exp(m_w - m*); since the
// final m* equals the exact segment max, gx = acc / (s + EPS) matches the
// reference's pyg_softmax + weighted pooling.
__global__ __launch_bounds__(128) void k_fused_attnpool(
        const float* __restrict__ x, const int* __restrict__ batch,
        const float* __restrict__ W, const float* __restrict__ bias,
        float* __restrict__ out, int N) {
    const int b = blockIdx.x;
    __shared__ float lds_w[DIM];
    __shared__ float lds_gx[DIM];
    __shared__ int   s_start, s_end;
    __shared__ float s_wm[4];
    __shared__ float s_den;

    const int tid = threadIdx.x;
    if (tid == 0) {
        // lower_bound(batch, b) / lower_bound(batch, b+1) on sorted batch
        int lo = 0, hi = N;
        while (lo < hi) { int m = (lo + hi) >> 1; if (batch[m] < b) lo = m + 1; else hi = m; }
        s_start = lo;
        int lo2 = lo, hi2 = N;
        while (lo2 < hi2) { int m = (lo2 + hi2) >> 1; if (batch[m] < b + 1) lo2 = m + 1; else hi2 = m; }
        s_end = lo2;
        s_den = 0.0f;
    }
    lds_w[tid]  = W[tid];      // blockDim.x == DIM
    lds_gx[tid] = 0.0f;
    __syncthreads();

    const int start = s_start, end = s_end, cnt = end - start;
    if (cnt == 0) {                      // uniform across block: safe early-out
        out[(size_t)b * DIM + tid] = 0.0f;
        return;
    }

    const float bb   = bias[0];
    const int   w    = tid >> 5;
    const int   lane = tid & 31;
    const int   n16  = lane & 15;
    const int   hi16 = lane >> 4;

    float m_w = NEG_INF;   // running segment max (wave-uniform)
    float s_l = 0.0f;      // per-lane slice of running exp-sum
    v8f   acc[8] = {};     // running weighted feature sums (WMMA C/D chain)

    const int nch = (cnt + 31) >> 5;
    const float4* w4 = (const float4*)lds_w;

    for (int c = w; c < nch; c += 4) {
        const int  cbase = start + (c << 5);
        const int  node  = cbase + lane;
        const bool valid = node < end;
        const int  rnode = valid ? node : start;

        // ---- 1. score for this lane's node: full-row dot against LDS-held W
        const float4* xr = (const float4*)(x + (size_t)rnode * DIM);
        float sc = bb;
        for (int j = 0; j < DIM / 4; ++j) {
            const float4 v  = xr[j];
            const float4 ww = w4[j];
            sc += v.x * ww.x + v.y * ww.y + v.z * ww.z + v.w * ww.w;
        }
        if (!valid) sc = NEG_INF;

        // ---- 2. online-softmax state update (wave-uniform rescale)
        float cm = sc;
#pragma unroll
        for (int o = 16; o; o >>= 1) cm = fmaxf(cm, __shfl_xor(cm, o, 32));
        const float m_new = fmaxf(m_w, cm);
        if (m_new > m_w) {               // wave-uniform branch, rare after warm-up
            const float f = (m_w == NEG_INF) ? 0.0f : __expf(m_w - m_new);
            s_l *= f;
#pragma unroll
            for (int g = 0; g < 8; ++g) acc[g] *= f;
            m_w = m_new;
        }

        const float p = valid ? __expf(sc - m_w) : 0.0f;
        s_l += p;

        // ---- 3. A matrix (16x32 f16, all rows == chunk attn weights)
        // 16-bit A layout: lanes 0-15 hold K={0..7,16..23}, lanes 16-31 K={8..15,24..31}
        v16h a;
#pragma unroll
        for (int e = 0; e < 16; ++e) {
            const int v = e >> 1, h = e & 1;
            const int k = (v < 4) ? (2 * v + h + 8 * hi16)
                                  : (16 + 2 * (v - 4) + h + 8 * hi16);
            const float t = __shfl(p, k, 32);
            a[e] = (_Float16)t;
        }

        // speculative prefetch of this wave's next chunk (global_prefetch_b8)
        if (c + 4 < nch) {
            size_t off  = (size_t)(cbase + 128) * DIM + (size_t)lane * 16;
            size_t last = (size_t)N * DIM - 1;
            if (off > last) off = last;
            __builtin_prefetch(x + off, 0, 1);
        }

        // ---- 4. 8 feature groups of 16 -> 8 chained WMMAs (L0 hits; rows warmed above)
#pragma unroll
        for (int g = 0; g < 8; ++g) {
            // 16-bit B layout (32x16): lanes 0-15 K={0..7,16..23}, lanes 16-31 K={8..15,24..31};
            // element e is K = (e&7) + 8*hi16 + 16*(e>>3), column N = lane&15.
            v16h bm;
#pragma unroll
            for (int e = 0; e < 16; ++e) {
                const int row = cbase + (e & 7) + 8 * hi16 + 16 * (e >> 3);
                const int rr  = (row < end) ? row : start;   // padded K has attn==0
                bm[e] = (_Float16)x[(size_t)rr * DIM + g * 16 + n16];
            }
            acc[g] = __builtin_amdgcn_wmma_f32_16x16x32_f16(
                false, a, false, bm, (short)0, acc[g], false, false);
        }
    }

    // ---- merge the 4 waves' online-softmax partials
    if (lane == 0) s_wm[w] = m_w;
    __syncthreads();
    const float mstar = fmaxf(fmaxf(s_wm[0], s_wm[1]), fmaxf(s_wm[2], s_wm[3]));
    const float scale = (m_w == NEG_INF) ? 0.0f : __expf(m_w - mstar);

    float st = s_l;
#pragma unroll
    for (int o = 16; o; o >>= 1) st += __shfl_xor(st, o, 32);
    if (lane == 0) atomicAdd(&s_den, st * scale);

    // rows of D are replicated -> element 0 carries feature g*16 + (lane&15)
#pragma unroll
    for (int g = 0; g < 8; ++g) {
        if (lane < 16) atomicAdd(&lds_gx[g * 16 + n16], acc[g][0] * scale);
    }
    __syncthreads();
    out[(size_t)b * DIM + tid] = lds_gx[tid] / (s_den + EPS);
}

extern "C" void kernel_launch(void* const* d_in, const int* in_sizes, int n_in,
                              void* d_out, int out_size, void* d_ws, size_t ws_size,
                              hipStream_t stream) {
    const float* x     = (const float*)d_in[0];
    // d_in[1] = edge_index (unused by the forward math)
    const int*   batch = (const int*)d_in[2];
    const float* W     = (const float*)d_in[3];
    const float* bias  = (const float*)d_in[4];
    const int N = in_sizes[2];       // batch has N elements
    const int B = out_size / DIM;    // gx is [B, 128]

    k_fused_attnpool<<<B, DIM, 0, stream>>>(x, batch, W, bias, (float*)d_out, N);
}
